// SwinFuseBlock_63513976373847
// MI455X (gfx1250) — matrix operand order repaired and checked by reading, not compiled, not run on hardware
//
#include <hip/hip_runtime.h>
#include <cstdint>

typedef __attribute__((ext_vector_type(16))) __bf16 v16bf;
typedef __attribute__((ext_vector_type(8)))  float  v8f;
typedef unsigned short u16;

// ---------------------------------------------------------------- helpers ---
static __device__ __forceinline__ float wave_sum(float x) {
#pragma unroll
  for (int off = 16; off > 0; off >>= 1) x += __shfl_xor(x, off, 32);
  return x;
}

static __device__ __forceinline__ v8f vzero8() {
  v8f z = {0.f, 0.f, 0.f, 0.f, 0.f, 0.f, 0.f, 0.f};
  return z;
}

// A-fragment (16x32 bf16, row-major source): lane holds row (lane&15),
// contiguous k = kbase + ((lane>=16)?16:0) + {0..15}.
static __device__ __forceinline__ v16bf load_a(const __bf16* row, int kbase, int lane) {
  const __bf16* p = row + kbase + ((lane & 16) ? 16 : 0);
  v16bf r;
#pragma unroll
  for (int i = 0; i < 16; i++) r[i] = p[i];
  return r;
}

// B-fragment for X @ W^T: B[k][n] = W[n][k].  lane holds column n = source row
// (lane&15); contiguous k = kbase + ((lane>=16)?16:0) + {0..15}.
static __device__ __forceinline__ v16bf load_bT(const __bf16* row, int kbase, int lane) {
  const __bf16* p = row + kbase + ((lane & 16) ? 16 : 0);
  v16bf r;
#pragma unroll
  for (int i = 0; i < 16; i++) r[i] = p[i];
  return r;
}

// B-fragment where B == source (row-major K x N): lane holds column n, strided
// reads down the column (used for P @ V).
static __device__ __forceinline__ v16bf load_b_col(const __bf16* colbase, int stride,
                                                   int kbase, int lane) {
  int k0 = kbase + ((lane & 16) ? 16 : 0);
  v16bf r;
#pragma unroll
  for (int i = 0; i < 16; i++) r[i] = colbase[(k0 + i) * stride];
  return r;
}

static __device__ __forceinline__ v8f wmma_bf(v16bf a, v16bf b, v8f c) {
  return __builtin_amdgcn_wmma_f32_16x16x32_bf16(false, a, false, b, (short)0, c, false, false);
}

static __device__ __forceinline__ float gelu_tanh(float x) {
  float x3 = x * x * x;
  return 0.5f * x * (1.0f + tanhf(0.7978845608028654f * (x + 0.044715f * x3)));
}

// ------------------------------------------------------------ tiny kernels ---
__global__ __launch_bounds__(256) void cvt_bf16(const float* __restrict__ in,
                                                u16* __restrict__ out, int n) {
  int i = blockIdx.x * 256 + threadIdx.x;
  if (i < n) ((__bf16*)out)[i] = (__bf16)in[i];
}

// interleave feat1/feat2 -> (B,784,192) and apply cp layer_norm. 1 wave/token.
__global__ __launch_bounds__(256) void fuse_ln(const float* __restrict__ f1,
                                               const float* __restrict__ f2,
                                               const float* __restrict__ g,
                                               const float* __restrict__ b,
                                               float* __restrict__ XA) {
  int token = blockIdx.x * 8 + (threadIdx.x >> 5);
  int lane = threadIdx.x & 31;
  int bb = token / 784, l = token % 784;
  int h = l / 28, w = l % 28;
  int i = h >> 1, r = h & 1, j = w >> 1, s = w & 1;
  const float* src = (r == 0) ? (s == 0 ? f1 : f2) : (s == 0 ? f2 : f1);
  float buf[6];
  float sum = 0.f, sum2 = 0.f;
  int idx = 0;
  for (int d = lane; d < 192; d += 32) {
    float v = src[(((size_t)bb * 384 + 2 * d + r) * 14 + i) * 14 + j];
    buf[idx++] = v; sum += v; sum2 += v * v;
  }
  sum = wave_sum(sum); sum2 = wave_sum(sum2);
  float mu = sum / 192.f;
  float rinv = rsqrtf(sum2 / 192.f - mu * mu + 1e-5f);
  idx = 0;
  for (int d = lane; d < 192; d += 32)
    XA[(size_t)token * 192 + d] = (buf[idx++] - mu) * rinv * g[d] + b[d];
}

// final transpose (B,196,384) -> (B,384,14,14)
__global__ __launch_bounds__(256) void out_tr(const float* __restrict__ X,
                                              float* __restrict__ O) {
  int idx = blockIdx.x * 256 + threadIdx.x;
  int b = idx / (384 * 196), rem = idx % (384 * 196);
  int c = rem / 196, p = rem % 196;
  O[idx] = X[((size_t)b * 196 + p) * 384 + c];
}

// --------------------------------------------------- fused window attention ---
// One workgroup (8 waves) per (batch, window). LN1 -> qkv GEMM -> per-head
// softmax(QK^T*scale + rpb + shift mask) V -> proj + residual.
template <int C, int HEADS, int H>
__global__ __launch_bounds__(256) void attn_kernel(
    const float* __restrict__ X, float* __restrict__ Y,
    const float* __restrict__ n1g, const float* __restrict__ n1b,
    const u16* __restrict__ Wqkv_, const float* __restrict__ bqkv,
    const u16* __restrict__ Wproj_, const float* __restrict__ bproj,
    const float* __restrict__ rpb, int shift) {
  constexpr int W = H;
  constexpr int NW = H / 7;
  constexpr int L = H * W;
  constexpr int N3C = 3 * C;
  const __bf16* Wqkv = (const __bf16*)Wqkv_;
  const __bf16* Wproj = (const __bf16*)Wproj_;
  const int lane = threadIdx.x & 31;
  const int wave = threadIdx.x >> 5;
  const int wid = blockIdx.x % (NW * NW);
  const int bb = blockIdx.x / (NW * NW);
  const int wy = wid / NW, wx = wid % NW;

  extern __shared__ char smem[];
  __bf16* lds_x = (__bf16*)smem;               // 64*C
  __bf16* lds_qkv = lds_x + 64 * C;            // 3*HEADS*64*32 = 192*C
  __bf16* lds_o = lds_qkv + 192 * C;           // 64*C
  float* lds_s = (float*)(lds_o + 64 * C);     // 8 waves * 16 * 64
  __bf16* lds_p = (__bf16*)(lds_s + 8 * 16 * 64);  // 8 waves * 16 * 64

  const float* xb = X + (size_t)bb * L * C;
  float* yb = Y + (size_t)bb * L * C;

  auto tok_l = [&](int t) -> int {  // window token -> image row index
    int ty = t / 7, tx = t % 7;
    int h = wy * 7 + ty + shift; if (h >= H) h -= H;
    int w = wx * 7 + tx + shift; if (w >= W) w -= W;
    return h * W + w;
  };

  // ---- phase 1: LN1 into lds_x (rows 49..63 zero) ----
  for (int t = wave; t < 64; t += 8) {
    if (t < 49) {
      const float* row = xb + (size_t)tok_l(t) * C;
      float s = 0.f, s2 = 0.f;
      for (int c = lane; c < C; c += 32) { float v = row[c]; s += v; s2 += v * v; }
      s = wave_sum(s); s2 = wave_sum(s2);
      float mu = s / C;
      float rinv = rsqrtf(s2 / C - mu * mu + 1e-5f);
      for (int c = lane; c < C; c += 32)
        lds_x[t * C + c] = (__bf16)((row[c] - mu) * rinv * n1g[c] + n1b[c]);
    } else {
      for (int c = lane; c < C; c += 32) lds_x[t * C + c] = (__bf16)0.f;
    }
  }
  __syncthreads();

  // ---- phase 2: qkv = lnx @ Wqkv^T + b ----
  constexpr float scale = 0.17677669529663687f;  // 1/sqrt(hd=32)
  {
    const int NT = N3C / 16;
    for (int tile = wave; tile < 4 * NT; tile += 8) {
      int mt = tile / NT, nt = tile % NT;
      int arow = mt * 16 + (lane & 15);
      int brow = nt * 16 + (lane & 15);
      v8f acc = vzero8();
      for (int ks = 0; ks < C / 32; ks++) {
        v16bf a = load_a(lds_x + arow * C, ks * 32, lane);
        v16bf b = load_bT(Wqkv + (size_t)brow * C, ks * 32, lane);
        acc = wmma_bf(a, b, acc);
      }
      int n = nt * 16 + (lane & 15);
      int which = n / C, c = n % C;
      int head = c >> 5, col = c & 31;
      float bias = bqkv[n];
      __bf16* dst = lds_qkv + (size_t)(which * HEADS + head) * 64 * 32 + col;
      int rbase = mt * 16 + ((lane & 16) ? 8 : 0);
#pragma unroll
      for (int v = 0; v < 8; v++) {
        float val = acc[v] + bias;
        if (which == 0) val *= scale;
        dst[(rbase + v) * 32] = (__bf16)val;
      }
    }
  }
  __syncthreads();

  // ---- phase 3: per (head, m-tile) attention unit ----
  for (int unit = wave; unit < HEADS * 4; unit += 8) {
    int head = unit >> 2, mt = unit & 3;
    const __bf16* Q = lds_qkv + (size_t)(0 * HEADS + head) * 64 * 32;
    const __bf16* K = lds_qkv + (size_t)(1 * HEADS + head) * 64 * 32;
    const __bf16* V = lds_qkv + (size_t)(2 * HEADS + head) * 64 * 32;
    float* S = lds_s + wave * 16 * 64;
    __bf16* P = lds_p + wave * 16 * 64;

    int arow = mt * 16 + (lane & 15);
    v16bf aq = load_a(Q + arow * 32, 0, lane);
#pragma unroll
    for (int nt = 0; nt < 4; nt++) {
      v16bf bk = load_bT(K + (size_t)(nt * 16 + (lane & 15)) * 32, 0, lane);
      v8f acc = vzero8();
      acc = wmma_bf(aq, bk, acc);
      int rb = (lane & 16) ? 8 : 0;
      int colp = nt * 16 + (lane & 15);
#pragma unroll
      for (int v = 0; v < 8; v++) S[(rb + v) * 64 + colp] = acc[v];
    }

    // row softmax: 2 lanes per row, 32 cols each
    int r = lane >> 1, half = lane & 1;
    int i = mt * 16 + r;
    int iy = i / 7, ix = i % 7;
    float vals[32];
    float mx = -3e38f;
#pragma unroll
    for (int j2 = 0; j2 < 32; j2++) {
      int j = half * 32 + j2;
      float v = S[r * 64 + j];
      if (j < 49 && i < 49) {
        int jy = j / 7, jx = j % 7;
        int ridx = (iy - jy + 6) * 13 + (ix - jx + 6);
        v += rpb[ridx * HEADS + head];
        if (shift > 0) {
          int ri = wy * 7 + iy, ci = wx * 7 + ix;
          int rj = wy * 7 + jy, cj = wx * 7 + jx;
          int regi = ((ri < H - 7) ? 0 : (ri < H - shift ? 1 : 2)) * 3 +
                     ((ci < W - 7) ? 0 : (ci < W - shift ? 1 : 2));
          int regj = ((rj < H - 7) ? 0 : (rj < H - shift ? 1 : 2)) * 3 +
                     ((cj < W - 7) ? 0 : (cj < W - shift ? 1 : 2));
          if (regi != regj) v -= 100.0f;
        }
      } else {
        v = -1e30f;
      }
      vals[j2] = v;
      mx = fmaxf(mx, v);
    }
    mx = fmaxf(mx, __shfl_xor(mx, 1, 32));
    float sum = 0.f;
#pragma unroll
    for (int j2 = 0; j2 < 32; j2++) { float e = __expf(vals[j2] - mx); vals[j2] = e; sum += e; }
    sum += __shfl_xor(sum, 1, 32);
    float inv = 1.0f / sum;
#pragma unroll
    for (int j2 = 0; j2 < 32; j2++) P[r * 64 + half * 32 + j2] = (__bf16)(vals[j2] * inv);

    // O = P @ V  (16x64 @ 64x32)
#pragma unroll
    for (int nt2 = 0; nt2 < 2; nt2++) {
      v8f acc = vzero8();
#pragma unroll
      for (int kb = 0; kb < 2; kb++) {
        v16bf pa = load_a(P + (lane & 15) * 64, kb * 32, lane);
        v16bf vb = load_b_col(V + nt2 * 16 + (lane & 15), 32, kb * 32, lane);
        acc = wmma_bf(pa, vb, acc);
      }
      int rbase = mt * 16 + ((lane & 16) ? 8 : 0);
      int col = head * 32 + nt2 * 16 + (lane & 15);
#pragma unroll
      for (int v = 0; v < 8; v++) lds_o[(rbase + v) * C + col] = (__bf16)acc[v];
    }
  }
  __syncthreads();

  // ---- phase 4: proj + residual ----
  {
    const int NT = C / 16;
    for (int tile = wave; tile < 4 * NT; tile += 8) {
      int mt = tile / NT, nt = tile % NT;
      int arow = mt * 16 + (lane & 15);
      int brow = nt * 16 + (lane & 15);
      v8f acc = vzero8();
      for (int ks = 0; ks < C / 32; ks++) {
        v16bf a = load_a(lds_o + arow * C, ks * 32, lane);
        v16bf b = load_bT(Wproj + (size_t)brow * C, ks * 32, lane);
        acc = wmma_bf(a, b, acc);
      }
      int n = nt * 16 + (lane & 15);
      float bias = bproj[n];
      int rbase = mt * 16 + ((lane & 16) ? 8 : 0);
#pragma unroll
      for (int v = 0; v < 8; v++) {
        int m = rbase + v;
        if (m < 49) {
          int l = tok_l(m);
          yb[(size_t)l * C + n] = xb[(size_t)l * C + n] + acc[v] + bias;
        }
      }
    }
  }
}

// ----------------------------------------------------------------- MLP ------
// One workgroup per 32 token rows. LN2 -> fc1+GELU (LDS) -> fc2 + residual.
template <int C>
__global__ __launch_bounds__(256) void mlp_kernel(
    const float* __restrict__ X, float* __restrict__ Y,
    const float* __restrict__ g2, const float* __restrict__ b2n,
    const u16* __restrict__ W1_, const float* __restrict__ b1,
    const u16* __restrict__ W2_, const float* __restrict__ b2) {
  constexpr int C4 = 4 * C;
  const __bf16* W1 = (const __bf16*)W1_;
  const __bf16* W2 = (const __bf16*)W2_;
  int lane = threadIdx.x & 31, wave = threadIdx.x >> 5;
  int r0 = blockIdx.x * 32;
  extern __shared__ char smem[];
  __bf16* lds_x = (__bf16*)smem;   // 32*C
  __bf16* lds_h = lds_x + 32 * C;  // 32*C4

  for (int t = wave; t < 32; t += 8) {
    const float* row = X + (size_t)(r0 + t) * C;
    float s = 0.f, s2 = 0.f;
    for (int c = lane; c < C; c += 32) { float v = row[c]; s += v; s2 += v * v; }
    s = wave_sum(s); s2 = wave_sum(s2);
    float mu = s / C;
    float rinv = rsqrtf(s2 / C - mu * mu + 1e-5f);
    for (int c = lane; c < C; c += 32)
      lds_x[t * C + c] = (__bf16)((row[c] - mu) * rinv * g2[c] + b2n[c]);
  }
  __syncthreads();

  {  // fc1 + gelu
    const int NT = C4 / 16;
    for (int tile = wave; tile < 2 * NT; tile += 8) {
      int mt = tile / NT, nt = tile % NT;
      int arow = mt * 16 + (lane & 15);
      int brow = nt * 16 + (lane & 15);
      v8f acc = vzero8();
      for (int ks = 0; ks < C / 32; ks++) {
        v16bf a = load_a(lds_x + arow * C, ks * 32, lane);
        v16bf b = load_bT(W1 + (size_t)brow * C, ks * 32, lane);
        acc = wmma_bf(a, b, acc);
      }
      int n = nt * 16 + (lane & 15);
      float bias = b1[n];
      int rbase = mt * 16 + ((lane & 16) ? 8 : 0);
#pragma unroll
      for (int v = 0; v < 8; v++)
        lds_h[(rbase + v) * C4 + n] = (__bf16)gelu_tanh(acc[v] + bias);
    }
  }
  __syncthreads();

  {  // fc2 + residual
    const int NT = C / 16;
    for (int tile = wave; tile < 2 * NT; tile += 8) {
      int mt = tile / NT, nt = tile % NT;
      int arow = mt * 16 + (lane & 15);
      int brow = nt * 16 + (lane & 15);
      v8f acc = vzero8();
      for (int ks = 0; ks < C4 / 32; ks++) {
        v16bf a = load_a(lds_h + arow * C4, ks * 32, lane);
        v16bf b = load_bT(W2 + (size_t)brow * C4, ks * 32, lane);
        acc = wmma_bf(a, b, acc);
      }
      int n = nt * 16 + (lane & 15);
      float bias = b2[n];
      int rbase = mt * 16 + ((lane & 16) ? 8 : 0);
#pragma unroll
      for (int v = 0; v < 8; v++) {
        int m = r0 + rbase + v;
        Y[(size_t)m * C + n] = X[(size_t)m * C + n] + acc[v] + bias;
      }
    }
  }
}

// -------------------------------------------------------- patch merge ------
__global__ __launch_bounds__(256) void pm_merge_ln(const float* __restrict__ X,
                                                   u16* __restrict__ XC,
                                                   const float* __restrict__ g,
                                                   const float* __restrict__ b) {
  int token = blockIdx.x * 8 + (threadIdx.x >> 5);
  int lane = threadIdx.x & 31;
  int bb = token / 196, t = token % 196;
  int i = t / 14, j = t % 14;
  const float* xb = X + (size_t)bb * 784 * 192;
  const float* seg[4] = {xb + ((2 * i) * 28 + 2 * j) * 192,
                         xb + ((2 * i + 1) * 28 + 2 * j) * 192,
                         xb + ((2 * i) * 28 + 2 * j + 1) * 192,
                         xb + ((2 * i + 1) * 28 + 2 * j + 1) * 192};
  float s = 0.f, s2 = 0.f;
  for (int q = 0; q < 4; q++)
    for (int c = lane; c < 192; c += 32) { float v = seg[q][c]; s += v; s2 += v * v; }
  s = wave_sum(s); s2 = wave_sum(s2);
  float mu = s / 768.f;
  float rinv = rsqrtf(s2 / 768.f - mu * mu + 1e-5f);
  __bf16* out = (__bf16*)XC + (size_t)token * 768;
  for (int q = 0; q < 4; q++)
    for (int c = lane; c < 192; c += 32) {
      int cc = q * 192 + c;
      out[cc] = (__bf16)((seg[q][c] - mu) * rinv * g[cc] + b[cc]);
    }
}

__global__ __launch_bounds__(256) void pm_gemm(const u16* __restrict__ A_,
                                               const u16* __restrict__ W_,
                                               float* __restrict__ Y) {
  const __bf16* A = (const __bf16*)A_;
  const __bf16* Wr = (const __bf16*)W_;
  int lane = threadIdx.x & 31, wave = threadIdx.x >> 5;
  int r0 = blockIdx.x * 64;
  const int NT = 384 / 16;
  for (int tile = wave; tile < 4 * NT; tile += 8) {
    int mt = tile / NT, nt = tile % NT;
    int arow = r0 + mt * 16 + (lane & 15);
    int brow = nt * 16 + (lane & 15);
    v8f acc = vzero8();
    for (int ks = 0; ks < 768 / 32; ks++) {
      v16bf a = load_a(A + (size_t)arow * 768, ks * 32, lane);
      v16bf b = load_bT(Wr + (size_t)brow * 768, ks * 32, lane);
      acc = wmma_bf(a, b, acc);
    }
    int n = nt * 16 + (lane & 15);
    int rbase = r0 + mt * 16 + ((lane & 16) ? 8 : 0);
#pragma unroll
    for (int v = 0; v < 8; v++) Y[(size_t)(rbase + v) * 384 + n] = acc[v];
  }
}

// ----------------------------------------------------------------- host -----
extern "C" void kernel_launch(void* const* d_in, const int* in_sizes, int n_in,
                              void* d_out, int out_size, void* d_ws, size_t ws_size,
                              hipStream_t stream) {
  (void)in_sizes; (void)n_in; (void)out_size; (void)ws_size;
  // pytree (sorted-key) input order:
  // 0 feat1, 1 feat2, 2 cp_norm_b, 3 cp_norm_g, 4..16 cpab[0], 17..29 cpab[1],
  // 30 pm_norm_b, 31 pm_norm_g, 32 pm_red_w, 33..45 pmab[0], 46..58 pmab[1]
  // block dict order: fc1_b, fc1_w, fc2_b, fc2_w, n1b, n1g, n2b, n2g,
  //                   proj_b, proj_w, qkv_b, qkv_w, rpb
  char* wsb = (char*)d_ws;
  const size_t SZ_XA = (size_t)256 * 784 * 192 * sizeof(float);
  float* XA = (float*)wsb;
  float* XB = (float*)(wsb + SZ_XA);
  u16* WREG = (u16*)(wsb + 2 * SZ_XA);
  size_t woff = 0;

  auto cvt = [&](int idx, size_t n) -> const u16* {
    u16* dst = WREG + woff;
    cvt_bf16<<<dim3((unsigned)((n + 255) / 256)), dim3(256), 0, stream>>>(
        (const float*)d_in[idx], dst, (int)n);
    woff += n;
    return dst;
  };

  struct BPtr {
    const float *fc1_b, *fc2_b, *n1b, *n1g, *n2b, *n2g, *proj_b, *qkv_b, *rpb;
    const u16 *fc1_w, *fc2_w, *proj_w, *qkv_w;
  };
  auto getb = [&](int ib, int Cb) -> BPtr {
    BPtr p;
    p.fc1_b = (const float*)d_in[ib + 0];
    p.fc2_b = (const float*)d_in[ib + 2];
    p.n1b = (const float*)d_in[ib + 4];
    p.n1g = (const float*)d_in[ib + 5];
    p.n2b = (const float*)d_in[ib + 6];
    p.n2g = (const float*)d_in[ib + 7];
    p.proj_b = (const float*)d_in[ib + 8];
    p.qkv_b = (const float*)d_in[ib + 10];
    p.rpb = (const float*)d_in[ib + 12];
    p.fc1_w = cvt(ib + 1, (size_t)4 * Cb * Cb);
    p.fc2_w = cvt(ib + 3, (size_t)4 * Cb * Cb);
    p.proj_w = cvt(ib + 9, (size_t)Cb * Cb);
    p.qkv_w = cvt(ib + 11, (size_t)3 * Cb * Cb);
    return p;
  };

  BPtr b10 = getb(4, 192), b11 = getb(17, 192);
  BPtr b20 = getb(33, 384), b21 = getb(46, 384);
  const u16* pm_red = cvt(32, (size_t)384 * 768);

  // interleave + cp layernorm
  fuse_ln<<<200704 / 8, 256, 0, stream>>>((const float*)d_in[0], (const float*)d_in[1],
                                          (const float*)d_in[3], (const float*)d_in[2], XA);

  // stage 1 (C=192, heads=6, H=28), shared-mem sizes per derivation
  const size_t shA1 = (size_t)640 * 192 + 49152;   // 172032
  const size_t shM1 = (size_t)320 * 192;           // 61440
  attn_kernel<192, 6, 28><<<256 * 16, 256, shA1, stream>>>(
      XA, XB, b10.n1g, b10.n1b, b10.qkv_w, b10.qkv_b, b10.proj_w, b10.proj_b, b10.rpb, 0);
  mlp_kernel<192><<<200704 / 32, 256, shM1, stream>>>(
      XB, XA, b10.n2g, b10.n2b, b10.fc1_w, b10.fc1_b, b10.fc2_w, b10.fc2_b);
  attn_kernel<192, 6, 28><<<256 * 16, 256, shA1, stream>>>(
      XA, XB, b11.n1g, b11.n1b, b11.qkv_w, b11.qkv_b, b11.proj_w, b11.proj_b, b11.rpb, 3);
  mlp_kernel<192><<<200704 / 32, 256, shM1, stream>>>(
      XB, XA, b11.n2g, b11.n2b, b11.fc1_w, b11.fc1_b, b11.fc2_w, b11.fc2_b);

  // patch merge: gather+LN (bf16 into XB region), then 768->384 WMMA GEMM
  pm_merge_ln<<<50176 / 8, 256, 0, stream>>>(XA, (u16*)XB, (const float*)d_in[31],
                                             (const float*)d_in[30]);
  pm_gemm<<<50176 / 64, 256, 0, stream>>>((const u16*)XB, pm_red, XA);

  // stage 2 (C=384, heads=12, H=14)
  const size_t shA2 = (size_t)640 * 384 + 49152;   // 294912
  const size_t shM2 = (size_t)320 * 384;           // 122880
  attn_kernel<384, 12, 14><<<256 * 4, 256, shA2, stream>>>(
      XA, XB, b20.n1g, b20.n1b, b20.qkv_w, b20.qkv_b, b20.proj_w, b20.proj_b, b20.rpb, 0);
  mlp_kernel<384><<<50176 / 32, 256, shM2, stream>>>(
      XB, XA, b20.n2g, b20.n2b, b20.fc1_w, b20.fc1_b, b20.fc2_w, b20.fc2_b);
  attn_kernel<384, 12, 14><<<256 * 4, 256, shA2, stream>>>(
      XA, XB, b21.n1g, b21.n1b, b21.qkv_w, b21.qkv_b, b21.proj_w, b21.proj_b, b21.rpb, 3);
  mlp_kernel<384><<<50176 / 32, 256, shM2, stream>>>(
      XB, XA, b21.n2g, b21.n2b, b21.fc1_w, b21.fc1_b, b21.fc2_w, b21.fc2_b);

  // (B,196,384) -> (B,384,14,14)
  out_tr<<<19267584 / 256, 256, 0, stream>>>(XA, (float*)d_out);
}
